// CGMessageBlock_38732015075502
// MI455X (gfx1250) — matrix operand order repaired and checked by reading, not compiled, not run on hardware
//
#include <hip/hip_runtime.h>
#include <hip/hip_bf16.h>
#include <math.h>

typedef __attribute__((ext_vector_type(16))) _Float16 v16h;
typedef __attribute__((ext_vector_type(8)))  float    v8f;

constexpr int BB = 4;
constexpr int NN = 192;
constexpr int DD = 128;
constexpr int RR = 32;
constexpr int RS = 40;    // padded LDS stride (halves) for rbf rows
constexpr int HS = 136;   // padded LDS stride (halves) for 128-wide rows

struct MainParams {
  const float* node_s;
  const float* node_v;
  const float* rbf;
  const float* r_hat;
  const float* mask;
  const float* b1[4];
  const float* b2[4];
  const float* ln_g;
  const float* ln_b;
  const float* Wo;
  const float* bo;
  const float* v_scale;
  const float* t_scale;
  const _Float16* W1t;   // [4][128][32]  (transposed: [d][r])
  const _Float16* W2t;   // [4][128][128] (transposed: [dout][din])
  float* out;
};

struct PrepParams {
  const float* W1[4];
  const float* W2[4];
  _Float16* W1t;
  _Float16* W2t;
};

// Convert radial-MLP weights to f16, transposed so WMMA B-fragments are
// contiguous per lane (lane -> N column, elements -> K).
__global__ void __launch_bounds__(256) cg_prep_weights(PrepParams P) {
  int idx = blockIdx.x * 256 + threadIdx.x;
  if (idx < 4 * DD * RR) {                       // W1t: [mi][d][r]
    int mi = idx >> 12, rem = idx & 4095;
    int d = rem >> 5, r = rem & 31;
    P.W1t[idx] = (_Float16)P.W1[mi][r * DD + d];
  }
  int i2 = idx - 4 * DD * RR;
  if (i2 >= 0 && i2 < 4 * DD * DD) {             // W2t: [mi][dout][din]
    int mi = i2 >> 14, rem = i2 & 16383;
    int dout = rem >> 7, din = rem & 127;
    P.W2t[i2] = (_Float16)P.W2[mi][din * DD + dout];
  }
}

// A-fragment (16x32 f16, M x K) from row-major LDS tile.
// Layout: lane&15 -> M; lane>>4 selects K range {0..15} vs {16..31} (matches frag_b_t).
__device__ inline v16h frag_a_lds(const _Float16* src, int row0, int k0, int stride, int lane) {
  int m  = lane & 15;
  int kb = k0 + ((lane >> 4) << 4);
  const _Float16* p = src + (row0 + m) * stride + kb;
  v16h a;
#pragma unroll
  for (int e = 0; e < 8; ++e) { a[e] = p[e]; a[8 + e] = p[8 + e]; }
  return a;
}

// B-fragment (32x16 f16, K x N) from an N-major (transposed) weight matrix.
// lane&15 -> N; lane>>4 selects K range base; 16 contiguous halves per lane.
__device__ inline v16h frag_b_t(const _Float16* wt, int k0, int n0, int stride, int lane) {
  int n  = n0 + (lane & 15);
  int kb = k0 + ((lane >> 4) << 4);
  const _Float16* p = wt + n * stride + kb;
  v16h b;
#pragma unroll
  for (int e = 0; e < 16; ++e) b[e] = p[e];
  return b;
}

// Fast silu: x * sigmoid(x) with single-instruction v_rcp_f32 (result is
// rounded to f16 anyway; full IEEE division is wasted VALU between WMMAs).
__device__ inline float fast_silu(float v) {
  return v * __builtin_amdgcn_rcpf(1.0f + __expf(-v));
}

__global__ void __launch_bounds__(256) cg_message_kernel(MainParams P) {
  __shared__ _Float16 s_rbf[NN * RS];      // rbf[b,i] as f16, padded stride
  __shared__ _Float16 s_H[NN * HS];        // silu(rbf@W1+b1) f16
  __shared__ _Float16 s_rad[NN * HS];      // radial MLP output f16
  __shared__ float s_Y1[NN * 4];
  __shared__ float s_Y2[NN * 8];
  __shared__ float s_mask[NN];
  __shared__ float s_nv[3 * DD];
  __shared__ float s_msg[DD];
  __shared__ float s_msgv[DD];
  __shared__ float s_dv[3 * DD];
  __shared__ float s_dt[5 * DD];
  __shared__ float s_red[5 * 256];
  __shared__ float s_x[DD];

  const int tid  = threadIdx.x;
  const int lane = tid & 31;
  const int wave = tid >> 5;
  const int bi   = blockIdx.x;      // bi = b*NN + i
  const int b    = bi / NN;

  __builtin_prefetch(P.W2t, 0, 3);  // pull f16 weights toward L2 (global_prefetch)

  // ---- stage per-(b,i) inputs ----
  const float* rbfp = P.rbf + (size_t)bi * (NN * RR);
  for (int idx = tid; idx < NN * RR; idx += 256) {
    int j = idx >> 5, r = idx & 31;
    s_rbf[j * RS + r] = (_Float16)rbfp[idx];
  }
  for (int j = tid; j < NN; j += 256) {
    const float* rh = P.r_hat + ((size_t)bi * NN + j) * 3;
    float rx = rh[0], ry = rh[1], rz = rh[2];
    float inv = __builtin_amdgcn_rcpf(sqrtf(rx * rx + ry * ry + rz * rz) + 1e-12f);
    float x = rx * inv, y = ry * inv, z = rz * inv;
    const float C1  = 0.48860251190291992f;   // sqrt(3/(4pi))
    const float CY  = 0.28209479177387814f;   // 0.5/sqrt(pi)
    const float S15 = 3.8729833462074170f;    // sqrt(15)
    const float S5H = 1.1180339887498949f;    // 0.5*sqrt(5)
    s_Y1[j * 4 + 0] = C1 * x; s_Y1[j * 4 + 1] = C1 * y;
    s_Y1[j * 4 + 2] = C1 * z; s_Y1[j * 4 + 3] = 0.f;
    s_Y2[j * 8 + 0] = CY * S15 * x * y;
    s_Y2[j * 8 + 1] = CY * S15 * y * z;
    s_Y2[j * 8 + 2] = CY * S5H * (3.f * z * z - 1.f);
    s_Y2[j * 8 + 3] = CY * S15 * x * z;
    s_Y2[j * 8 + 4] = CY * 0.5f * S15 * (x * x - y * y);
    s_Y2[j * 8 + 5] = 0.f; s_Y2[j * 8 + 6] = 0.f; s_Y2[j * 8 + 7] = 0.f;
    s_mask[j] = P.mask[(size_t)bi * NN + j];
  }
  for (int idx = tid; idx < 3 * DD; idx += 256)
    s_nv[idx] = P.node_v[(size_t)bi * 3 * DD + idx];
  __syncthreads();

  const float* nsb = P.node_s + (size_t)b * NN * DD;
  const int d0   = wave * 16;            // this wave's 16-wide d column tile
  const int col  = d0 + (lane & 15);
  const int rhi  = (lane >> 4) << 3;     // accumulator row offset (0 or 8)
  const int dR   = tid & 127;            // reduction: feature index
  const int half = tid >> 7;             // reduction: j-half
  const int jbeg = half * 96;

  for (int mi = 0; mi < 4; ++mi) {
    // ---- GEMM1: H = silu(rbf @ W1 + b1), K = 32 (one WMMA per tile) ----
    {
      const _Float16* w1t = P.W1t + mi * (DD * RR);
      v16h bf = frag_b_t(w1t, 0, d0, RR, lane);
      float b1v = P.b1[mi][col];
      for (int jt = 0; jt < 12; ++jt) {
        v16h af = frag_a_lds(s_rbf, jt * 16, 0, RS, lane);
        v8f acc = {};
        acc = __builtin_amdgcn_wmma_f32_16x16x32_f16(false, af, false, bf,
                                                     (short)0, acc, false, false);
#pragma unroll
        for (int r = 0; r < 8; ++r) {
          float v = acc[r] + b1v;
          s_H[(jt * 16 + rhi + r) * HS + col] = (_Float16)fast_silu(v);
        }
      }
    }
    __syncthreads();
    // ---- GEMM2: rad = H @ W2 + b2, K = 128 (4 WMMA k-steps) ----
    {
      const _Float16* w2t = P.W2t + mi * (DD * DD);
      v16h bf0 = frag_b_t(w2t,  0, d0, DD, lane);
      v16h bf1 = frag_b_t(w2t, 32, d0, DD, lane);
      v16h bf2 = frag_b_t(w2t, 64, d0, DD, lane);
      v16h bf3 = frag_b_t(w2t, 96, d0, DD, lane);
      float b2v = P.b2[mi][col];
      for (int jt = 0; jt < 12; ++jt) {
        v8f acc = {};
        v16h a0 = frag_a_lds(s_H, jt * 16,  0, HS, lane);
        acc = __builtin_amdgcn_wmma_f32_16x16x32_f16(false, a0, false, bf0, (short)0, acc, false, false);
        v16h a1 = frag_a_lds(s_H, jt * 16, 32, HS, lane);
        acc = __builtin_amdgcn_wmma_f32_16x16x32_f16(false, a1, false, bf1, (short)0, acc, false, false);
        v16h a2 = frag_a_lds(s_H, jt * 16, 64, HS, lane);
        acc = __builtin_amdgcn_wmma_f32_16x16x32_f16(false, a2, false, bf2, (short)0, acc, false, false);
        v16h a3 = frag_a_lds(s_H, jt * 16, 96, HS, lane);
        acc = __builtin_amdgcn_wmma_f32_16x16x32_f16(false, a3, false, bf3, (short)0, acc, false, false);
#pragma unroll
        for (int r = 0; r < 8; ++r)
          s_rad[(jt * 16 + rhi + r) * HS + col] = (_Float16)(acc[r] + b2v);
      }
    }
    __syncthreads();
    // ---- j-reduction for this MLP (mask powers folded in) ----
    {
      float p0 = 0.f, p1 = 0.f, p2 = 0.f, p3 = 0.f, p4 = 0.f;
      if (mi == 0) {                       // msg_s: m^2 * w0 * node_s
        for (int j = jbeg; j < jbeg + 96; ++j) {
          float mk = s_mask[j];
          p0 += mk * mk * (float)s_rad[j * HS + dR] * nsb[j * DD + dR];
        }
      } else if (mi == 1) {                // msg_s_from_v: m * w110 * (v . Y1)
        float v0 = s_nv[dR], v1 = s_nv[DD + dR], v2 = s_nv[2 * DD + dR];
        for (int j = jbeg; j < jbeg + 96; ++j) {
          float dv = v0 * s_Y1[j * 4 + 0] + v1 * s_Y1[j * 4 + 1] + v2 * s_Y1[j * 4 + 2];
          p0 += s_mask[j] * dv * (float)s_rad[j * HS + dR];
        }
      } else if (mi == 2) {                // delta_v: m^2 * wv * s_j outer Y1
        for (int j = jbeg; j < jbeg + 96; ++j) {
          float mk = s_mask[j];
          float c = mk * mk * (float)s_rad[j * HS + dR] * nsb[j * DD + dR];
          p0 += c * s_Y1[j * 4 + 0];
          p1 += c * s_Y1[j * 4 + 1];
          p2 += c * s_Y1[j * 4 + 2];
        }
      } else {                             // delta_t: m^2 * wt * s_j outer Y2
        for (int j = jbeg; j < jbeg + 96; ++j) {
          float mk = s_mask[j];
          float c = mk * mk * (float)s_rad[j * HS + dR] * nsb[j * DD + dR];
          p0 += c * s_Y2[j * 8 + 0];
          p1 += c * s_Y2[j * 8 + 1];
          p2 += c * s_Y2[j * 8 + 2];
          p3 += c * s_Y2[j * 8 + 3];
          p4 += c * s_Y2[j * 8 + 4];
        }
      }
      s_red[tid] = p0;        s_red[256 + tid] = p1;  s_red[512 + tid] = p2;
      s_red[768 + tid] = p3;  s_red[1024 + tid] = p4;
    }
    __syncthreads();
    if (tid < DD) {
      float r0 = s_red[tid]        + s_red[128 + tid];
      float r1 = s_red[256 + tid]  + s_red[384 + tid];
      float r2 = s_red[512 + tid]  + s_red[640 + tid];
      float r3 = s_red[768 + tid]  + s_red[896 + tid];
      float r4 = s_red[1024 + tid] + s_red[1152 + tid];
      if      (mi == 0) s_msg[tid]  = r0;
      else if (mi == 1) s_msgv[tid] = r0;
      else if (mi == 2) { s_dv[tid] = r0; s_dv[DD + tid] = r1; s_dv[2 * DD + tid] = r2; }
      else { s_dt[tid] = r0; s_dt[DD + tid] = r1; s_dt[2 * DD + tid] = r2;
             s_dt[3 * DD + tid] = r3; s_dt[4 * DD + tid] = r4; }
    }
    __syncthreads();
  }

  // ---- LayerNorm(x) @ Wo + bo ----
  float xv = 0.f;
  if (tid < DD) xv = s_msg[tid] + s_msgv[tid];
  s_red[tid]       = (tid < DD) ? xv : 0.f;
  s_red[256 + tid] = (tid < DD) ? xv * xv : 0.f;
  __syncthreads();
  for (int s = 128; s > 0; s >>= 1) {
    if (tid < s) { s_red[tid] += s_red[tid + s]; s_red[256 + tid] += s_red[256 + tid + s]; }
    __syncthreads();
  }
  float mu   = s_red[0] * (1.0f / DD);
  float var  = s_red[256] * (1.0f / DD) - mu * mu;
  float rstd = rsqrtf(var + 1e-5f);
  if (tid < DD) s_x[tid] = (xv - mu) * rstd * P.ln_g[tid] + P.ln_b[tid];
  __syncthreads();
  if (tid < DD) {
    float acc = P.bo[tid];
    for (int k = 0; k < DD; ++k) acc += s_x[k] * P.Wo[k * DD + tid];
    P.out[(size_t)bi * DD + tid] = acc;
  }
  const size_t OFFV = (size_t)BB * NN * DD;
  const size_t OFFT = OFFV + (size_t)BB * NN * 3 * DD;
  for (int idx = tid; idx < 3 * DD; idx += 256) {
    int dd = idx & 127;
    P.out[OFFV + (size_t)bi * 3 * DD + idx] = s_dv[idx] * P.v_scale[dd];
  }
  for (int idx = tid; idx < 5 * DD; idx += 256) {
    int dd = idx & 127;
    P.out[OFFT + (size_t)bi * 5 * DD + idx] = s_dt[idx] * P.t_scale[dd];
  }
}

extern "C" void kernel_launch(void* const* d_in, const int* in_sizes, int n_in,
                              void* d_out, int out_size, void* d_ws, size_t ws_size,
                              hipStream_t stream) {
  (void)in_sizes; (void)n_in; (void)out_size; (void)ws_size;

  _Float16* W1t = reinterpret_cast<_Float16*>(d_ws);
  _Float16* W2t = W1t + 4 * DD * RR;

  PrepParams pp;
  pp.W1[0] = (const float*)d_in[6];  pp.W1[1] = (const float*)d_in[10];
  pp.W1[2] = (const float*)d_in[14]; pp.W1[3] = (const float*)d_in[18];
  pp.W2[0] = (const float*)d_in[8];  pp.W2[1] = (const float*)d_in[12];
  pp.W2[2] = (const float*)d_in[16]; pp.W2[3] = (const float*)d_in[20];
  pp.W1t = W1t; pp.W2t = W2t;
  int prep_elems = 4 * DD * RR + 4 * DD * DD;
  cg_prep_weights<<<(prep_elems + 255) / 256, 256, 0, stream>>>(pp);

  MainParams mp;
  mp.node_s = (const float*)d_in[0];
  mp.node_v = (const float*)d_in[1];
  mp.rbf    = (const float*)d_in[3];
  mp.r_hat  = (const float*)d_in[4];
  mp.mask   = (const float*)d_in[5];
  mp.b1[0] = (const float*)d_in[7];  mp.b1[1] = (const float*)d_in[11];
  mp.b1[2] = (const float*)d_in[15]; mp.b1[3] = (const float*)d_in[19];
  mp.b2[0] = (const float*)d_in[9];  mp.b2[1] = (const float*)d_in[13];
  mp.b2[2] = (const float*)d_in[17]; mp.b2[3] = (const float*)d_in[21];
  mp.ln_g = (const float*)d_in[22]; mp.ln_b = (const float*)d_in[23];
  mp.Wo   = (const float*)d_in[24]; mp.bo   = (const float*)d_in[25];
  mp.v_scale = (const float*)d_in[26];
  mp.t_scale = (const float*)d_in[27];
  mp.W1t = W1t; mp.W2t = W2t;
  mp.out = (float*)d_out;
  cg_message_kernel<<<BB * NN, 256, 0, stream>>>(mp);
}
